// Bottle2neck_40372692583008
// MI455X (gfx1250) — matrix operand order repaired and checked by reading, not compiled, not run on hardware
//
#include <hip/hip_runtime.h>

typedef __attribute__((ext_vector_type(16))) __bf16 v16bf;
typedef __attribute__((ext_vector_type(8)))  float  v8f;

// ---------------------------------------------------------------------------
// WMMA fragment loaders (bf16, 16x16x32). Per CDNA5 ISA 7.12.2:
//  A (16x32, row-major [M][K]): lane m=lane&15, hf=lane>>4 needs
//    K = k0+hf*8 .. +7   (16B)  and  K = k0+16+hf*8 .. +7 (16B)
//  B (32x16) from transposed weights Wt[N][K]: lane needs col N=m,
//    K = k0+hf*8 .. +7   (16B)  and  K = k0+16+hf*8 .. +7 (16B)
// ---------------------------------------------------------------------------
__device__ __forceinline__ v16bf load_a_frag(const __bf16* row_k0, int hf) {
  union { uint4 q[2]; v16bf v; } u;
  u.q[0] = *reinterpret_cast<const uint4*>(row_k0 + hf * 8);
  u.q[1] = *reinterpret_cast<const uint4*>(row_k0 + 16 + hf * 8);
  return u.v;
}
__device__ __forceinline__ v16bf load_b_frag(const __bf16* col_k0, int hf) {
  union { uint4 q[2]; v16bf v; } u;
  const __bf16* p = col_k0 + hf * 8;
  u.q[0] = *reinterpret_cast<const uint4*>(p);
  u.q[1] = *reinterpret_cast<const uint4*>(p + 16);
  return u.v;
}
#define WMMA_BF16(A, B, C) \
  __builtin_amdgcn_wmma_f32_16x16x32_bf16(false, (A), false, (B), (short)0, (C), false, false)

// ---------------------------------------------------------------------------
// Dense GEMM: C[rows][ldc] = A[rows][K] * Bt[cols][K]^T  (bf16 in, f32 out)
// wave tile = 16 rows x 64 cols; block = 2 waves (32 rows); grid.y = cols/64.
// Bt is padded so blockIdx.y*64+63 is always a valid row of Bt.
// ---------------------------------------------------------------------------
__global__ __launch_bounds__(64) void wmma_gemm_kernel(
    const __bf16* __restrict__ A, const __bf16* __restrict__ Bt,
    float* __restrict__ C, int K, int ldc) {
  const int lane = threadIdx.x & 31;
  const int wave = threadIdx.x >> 5;
  const int m    = lane & 15;
  const int hf   = lane >> 4;
  const int row0 = (blockIdx.x * 2 + wave) * 16;
  const int c0   = blockIdx.y * 64;

  v8f z = {0.f, 0.f, 0.f, 0.f, 0.f, 0.f, 0.f, 0.f};
  v8f acc0 = z, acc1 = z, acc2 = z, acc3 = z;

  const __bf16* Arow  = A  + (size_t)(row0 + m) * K;
  const __bf16* Bbase = Bt + (size_t)(c0 + m) * K;

  for (int k0 = 0; k0 < K; k0 += 32) {
    v16bf a  = load_a_frag(Arow + k0, hf);
    v16bf b0 = load_b_frag(Bbase + k0, hf);
    v16bf b1 = load_b_frag(Bbase + (size_t)16 * K + k0, hf);
    v16bf b2 = load_b_frag(Bbase + (size_t)32 * K + k0, hf);
    v16bf b3 = load_b_frag(Bbase + (size_t)48 * K + k0, hf);
    acc0 = WMMA_BF16(a, b0, acc0);
    acc1 = WMMA_BF16(a, b1, acc1);
    acc2 = WMMA_BF16(a, b2, acc2);
    acc3 = WMMA_BF16(a, b3, acc3);
  }
  // C/D layout: VGPR r -> row (row0 + hf*8 + r), col c0 + t*16 + m
  float* crow = C + (size_t)(row0 + hf * 8) * ldc + c0 + m;
#pragma unroll
  for (int r = 0; r < 8; ++r) {
    crow[(size_t)r * ldc + 0]  = acc0[r];
    crow[(size_t)r * ldc + 16] = acc1[r];
    crow[(size_t)r * ldc + 32] = acc2[r];
    crow[(size_t)r * ldc + 48] = acc3[r];
  }
}

// ---------------------------------------------------------------------------
// Gather-GEMM sparse conv: out[n][c] = sum_k sum_j spin[nbr[n][k]][j]*Wt[k][c][j]
// spin: [(N+1)][128] bf16 (row N = zero row, cols 104..127 zero-padded)
// Wt:   [27][112][128] bf16 (transposed + zero-padded weights)
// out:  [N][112] f32.
// Block = 10 waves (160 rows). Per kernel offset k, the 28.6KB weight tile
// Wk[112][128] is staged into LDS once per block and shared by all 10 waves
// (10x reduction in L2 weight traffic). Each wave computes 16 rows x 112 cols
// (7 accumulators), reusing each gathered A fragment across all 7 col tiles.
// ---------------------------------------------------------------------------
#define CONV_WAVES 10
__global__ __launch_bounds__(CONV_WAVES * 32) void wmma_sconv_kernel(
    const __bf16* __restrict__ spin, const int* __restrict__ nbr,
    const __bf16* __restrict__ Wt, float* __restrict__ out) {
  __shared__ __align__(16) __bf16 Blds[112 * 128];   // 28672 B
  const int tid  = threadIdx.x;
  const int lane = tid & 31;
  const int wave = tid >> 5;
  const int m    = lane & 15;
  const int hf   = lane >> 4;
  const int row0 = (blockIdx.x * CONV_WAVES + wave) * 16;
  const int row  = row0 + m;

  v8f z = {0.f, 0.f, 0.f, 0.f, 0.f, 0.f, 0.f, 0.f};
  v8f acc[7];
#pragma unroll
  for (int t = 0; t < 7; ++t) acc[t] = z;

  for (int k = 0; k < 27; ++k) {
    // ---- stage Wk into LDS (cooperative b128 copy) ----
    __syncthreads();                       // previous iteration's reads done
    {
      const uint4* src = reinterpret_cast<const uint4*>(Wt + (size_t)k * (112 * 128));
      uint4* dst = reinterpret_cast<uint4*>(Blds);
      for (int i = tid; i < (112 * 128 * 2) / 16; i += CONV_WAVES * 32) dst[i] = src[i];
    }
    __syncthreads();

    const int idx = nbr[(size_t)row * 27 + k];       // == N -> zero row
    const __bf16* Ar = spin + (size_t)idx * 128;
    const __bf16* Wl = Blds + (size_t)m * 128;
#pragma unroll
    for (int kk = 0; kk < 4; ++kk) {
      v16bf a = load_a_frag(Ar + kk * 32, hf);
#pragma unroll
      for (int t = 0; t < 7; ++t) {
        v16bf b = load_b_frag(Wl + (size_t)t * (16 * 128) + kk * 32, hf);
        acc[t] = WMMA_BF16(a, b, acc[t]);
      }
    }
  }
  float* crow = out + (size_t)(row0 + hf * 8) * 112 + m;
#pragma unroll
  for (int t = 0; t < 7; ++t) {
#pragma unroll
    for (int r = 0; r < 8; ++r) crow[(size_t)r * 112 + t * 16] = acc[t][r];
  }
}

// ---------------------------------------------------------------------------
// Column stats: per-channel sum / sumsq over rows (training-mode BN), atomics.
// ---------------------------------------------------------------------------
__global__ void colstats_kernel(const float* __restrict__ X, int nrows, int ncols,
                                int ldx, float* __restrict__ raw, int rpb) {
  const int r0 = blockIdx.x * rpb;
  const int r1 = min(r0 + rpb, nrows);
  for (int c = threadIdx.x; c < ncols; c += blockDim.x) {
    float s = 0.f, s2 = 0.f;
    for (int r = r0; r < r1; ++r) {
      float v = X[(size_t)r * ldx + c];
      s += v; s2 += v * v;
    }
    atomicAdd(&raw[c], s);
    atomicAdd(&raw[ncols + c], s2);
  }
}

__global__ void bnstats_kernel(const float* __restrict__ raw, const float* __restrict__ g,
                               const float* __restrict__ b, float* __restrict__ ab,
                               int ncols, float invN) {
  int c = blockIdx.x * blockDim.x + threadIdx.x;
  if (c >= ncols) return;
  float mean = raw[c] * invN;
  float var  = fmaxf(raw[ncols + c] * invN - mean * mean, 0.f);
  float a    = g[c] * rsqrtf(var + 1e-5f);
  ab[c]         = a;
  ab[ncols + c] = b[c] - mean * a;
}

__global__ void zero_kernel(float* __restrict__ p, int n) {
  int i = blockIdx.x * blockDim.x + threadIdx.x;
  if (i < n) p[i] = 0.f;
}

// ---------------------------------------------------------------------------
// Weight / activation conversion kernels (fp32 -> bf16, transpose + pad)
// ---------------------------------------------------------------------------
__global__ void f2bf_kernel(const float* __restrict__ in, __bf16* __restrict__ out,
                            long long n) {
  long long i = (long long)blockIdx.x * blockDim.x + threadIdx.x;
  if (i < n) out[i] = (__bf16)in[i];
}

__global__ void w1t_kernel(const float* __restrict__ W1, __bf16* __restrict__ W1t) {
  int i = blockIdx.x * blockDim.x + threadIdx.x;       // W1t [448][256]
  if (i >= 448 * 256) return;
  int c = i >> 8, k = i & 255;
  W1t[i] = (c < 416) ? (__bf16)W1[(size_t)k * 416 + c] : (__bf16)0.f;
}

__global__ void wct_kernel(const float* __restrict__ Wc, __bf16* __restrict__ Wct) {
  long long i = (long long)blockIdx.x * blockDim.x + threadIdx.x;  // [3][27][112][128]
  if (i >= (long long)3 * 27 * 112 * 128) return;
  int j = (int)(i & 127);
  long long t = i >> 7;
  int c = (int)(t % 112); t /= 112;
  int k = (int)(t % 27);
  int s = (int)(t / 27);
  float v = 0.f;
  if (c < 104 && j < 104) v = Wc[(((size_t)s * 27 + k) * 104 + j) * 104 + c];
  Wct[i] = (__bf16)v;
}

__global__ void w3t_kernel(const float* __restrict__ W3, __bf16* __restrict__ W3t) {
  int i = blockIdx.x * blockDim.x + threadIdx.x;       // W3t [256][416]
  if (i >= 256 * 416) return;
  int c = i / 416, k = i % 416;
  W3t[i] = (__bf16)W3[(size_t)k * 256 + c];
}

// ---------------------------------------------------------------------------
// Stage-prep elementwise kernels (apply folded BN scale/bias + ReLU)
// ---------------------------------------------------------------------------
__global__ void prep0_kernel(const float* __restrict__ pre1, const float* __restrict__ ab1,
                             __bf16* __restrict__ spin, int nrows) {
  long long i = (long long)blockIdx.x * blockDim.x + threadIdx.x;
  if (i >= (long long)(nrows + 1) * 128) return;
  int j = (int)(i & 127);
  long long r = i >> 7;
  __bf16 v = (__bf16)0.f;
  if (r < nrows && j < 104) {
    float x = pre1[r * 448 + j] * ab1[j] + ab1[416 + j];
    v = (__bf16)fmaxf(x, 0.f);
  }
  spin[i] = v;
}

__global__ void prep_mid_kernel(const float* __restrict__ convraw, const float* __restrict__ abc,
                                const float* __restrict__ pre1, const float* __restrict__ ab1,
                                __bf16* __restrict__ spin, __bf16* __restrict__ cat,
                                int slicecol, int partcol, int nrows) {
  long long i = (long long)blockIdx.x * blockDim.x + threadIdx.x;
  if (i >= (long long)(nrows + 1) * 128) return;
  int j = (int)(i & 127);
  long long r = i >> 7;
  if (r >= nrows || j >= 104) { spin[i] = (__bf16)0.f; return; }
  float sp = fmaxf(convraw[r * 112 + j] * abc[j] + abc[104 + j], 0.f);
  cat[r * 416 + partcol + j] = (__bf16)sp;
  float sx = fmaxf(pre1[r * 448 + slicecol + j] * ab1[slicecol + j]
                   + ab1[416 + slicecol + j], 0.f);
  spin[i] = (__bf16)(sp + sx);
}

__global__ void prep_last_kernel(const float* __restrict__ convraw, const float* __restrict__ abc,
                                 const float* __restrict__ pre1, const float* __restrict__ ab1,
                                 __bf16* __restrict__ cat, int nrows) {
  long long i = (long long)blockIdx.x * blockDim.x + threadIdx.x;
  if (i >= (long long)nrows * 104) return;
  int j = (int)(i % 104);
  long long r = i / 104;
  float sp = fmaxf(convraw[r * 112 + j] * abc[j] + abc[104 + j], 0.f);
  cat[r * 416 + 208 + j] = (__bf16)sp;
  float sx = fmaxf(pre1[r * 448 + 312 + j] * ab1[312 + j] + ab1[416 + 312 + j], 0.f);
  cat[r * 416 + 312 + j] = (__bf16)sx;
}

__global__ void final_kernel(const float* __restrict__ pre3, const float* __restrict__ ab3,
                             const float* __restrict__ x, float* __restrict__ out,
                             long long n) {
  long long i = (long long)blockIdx.x * blockDim.x + threadIdx.x;
  if (i >= n) return;
  int c = (int)(i & 255);
  out[i] = fmaxf(pre3[i] * ab3[c] + ab3[256 + c] + x[i], 0.f);
}

// ---------------------------------------------------------------------------
extern "C" void kernel_launch(void* const* d_in, const int* in_sizes, int n_in,
                              void* d_out, int out_size, void* d_ws, size_t ws_size,
                              hipStream_t stream) {
  const float* x_feat = (const float*)d_in[0];
  const int*   nbr    = (const int*)d_in[1];
  const float* W1     = (const float*)d_in[2];
  const float* gamma1 = (const float*)d_in[3];
  const float* beta1  = (const float*)d_in[4];
  const float* Wc     = (const float*)d_in[5];
  const float* gammac = (const float*)d_in[6];
  const float* betac  = (const float*)d_in[7];
  const float* W3     = (const float*)d_in[8];
  const float* gamma3 = (const float*)d_in[9];
  const float* beta3  = (const float*)d_in[10];
  float* out = (float*)d_out;
  const int N = in_sizes[0] / 256;                 // 100000: mult of 32 and 160

  // ---- workspace carve-out (256B aligned) ----
  char* ws = (char*)d_ws;
  size_t off = 0;
  auto carve = [&](size_t bytes) -> void* {
    void* p = ws + off;
    off += (bytes + 255) & ~(size_t)255;
    return p;
  };
  __bf16* x_bf    = (__bf16*)carve((size_t)N * 256 * 2);
  __bf16* W1t     = (__bf16*)carve((size_t)448 * 256 * 2);
  __bf16* Wct     = (__bf16*)carve((size_t)3 * 27 * 112 * 128 * 2);
  __bf16* W3t     = (__bf16*)carve((size_t)256 * 416 * 2);
  float*  pre1    = (float*)carve((size_t)N * 448 * 4);
  __bf16* spin    = (__bf16*)carve((size_t)(N + 1) * 128 * 2);
  float*  convraw = (float*)carve((size_t)N * 112 * 4);
  __bf16* cat     = (__bf16*)carve((size_t)N * 416 * 2);
  float*  pre3    = (float*)carve((size_t)N * 256 * 4);
  float*  raw     = (float*)carve((size_t)2 * 416 * 4);
  float*  ab1     = (float*)carve((size_t)2 * 416 * 4);
  float*  abc     = (float*)carve((size_t)2 * 104 * 4);
  float*  ab3     = (float*)carve((size_t)2 * 256 * 4);
  (void)ws_size; (void)n_in; (void)out_size;

  auto gs = [](long long n) { return (unsigned)((n + 255) / 256); };
  const float invN = 1.f / (float)N;
  const int rowBlk     = N / 32;                   // 3125 (2-wave GEMM blocks)
  const int rowBlkConv = N / (CONV_WAVES * 16);    // 625  (10-wave conv blocks)
  const int statBlk    = (N + 1023) / 1024;

  // ---- convert inputs/weights to bf16 (transpose + pad weights) ----
  f2bf_kernel<<<gs((long long)N * 256), 256, 0, stream>>>(x_feat, x_bf, (long long)N * 256);
  w1t_kernel<<<gs(448 * 256), 256, 0, stream>>>(W1, W1t);
  wct_kernel<<<gs((long long)3 * 27 * 112 * 128), 256, 0, stream>>>(Wc, Wct);
  w3t_kernel<<<gs(256 * 416), 256, 0, stream>>>(W3, W3t);

  // ---- stage 1: GEMM1 + BN stats ----
  wmma_gemm_kernel<<<dim3(rowBlk, 7), 64, 0, stream>>>(x_bf, W1t, pre1, 256, 448);
  zero_kernel<<<4, 256, 0, stream>>>(raw, 832);
  colstats_kernel<<<statBlk, 256, 0, stream>>>(pre1, N, 416, 448, raw, 1024);
  bnstats_kernel<<<1, 512, 0, stream>>>(raw, gamma1, beta1, ab1, 416, invN);
  prep0_kernel<<<gs((long long)(N + 1) * 128), 256, 0, stream>>>(pre1, ab1, spin, N);

  // ---- 3 chained gather-GEMM sparse convs ----
  for (int i = 0; i < 3; ++i) {
    wmma_sconv_kernel<<<rowBlkConv, CONV_WAVES * 32, 0, stream>>>(
        spin, nbr, Wct + (size_t)i * 27 * 112 * 128, convraw);
    zero_kernel<<<4, 256, 0, stream>>>(raw, 832);
    colstats_kernel<<<statBlk, 256, 0, stream>>>(convraw, N, 104, 112, raw, 1024);
    bnstats_kernel<<<1, 512, 0, stream>>>(raw, gammac + i * 104, betac + i * 104,
                                          abc, 104, invN);
    if (i < 2) {
      prep_mid_kernel<<<gs((long long)(N + 1) * 128), 256, 0, stream>>>(
          convraw, abc, pre1, ab1, spin, cat, (i + 1) * 104, i * 104, N);
    } else {
      prep_last_kernel<<<gs((long long)N * 104), 256, 0, stream>>>(
          convraw, abc, pre1, ab1, cat, N);
    }
  }

  // ---- stage 3: GEMM3 + BN stats + residual/ReLU ----
  wmma_gemm_kernel<<<dim3(rowBlk, 4), 64, 0, stream>>>(cat, W3t, pre3, 416, 256);
  zero_kernel<<<4, 256, 0, stream>>>(raw, 832);
  colstats_kernel<<<statBlk, 256, 0, stream>>>(pre3, N, 256, 256, raw, 1024);
  bnstats_kernel<<<1, 512, 0, stream>>>(raw, gamma3, beta3, ab3, 256, invN);
  final_kernel<<<gs((long long)N * 256), 256, 0, stream>>>(
      pre3, ab3, x_feat, out, (long long)N * 256);
}
